// MultiHeadDiffNetPlusPlus_53635551592546
// MI455X (gfx1250) — matrix operand order repaired and checked by reading, not compiled, not run on hardware
//
#include <hip/hip_runtime.h>
#include <hip/hip_bf16.h>

// MultiHeadDiffNet++ forward, MI455X (gfx1250).
// - K/V projections + fusion/help GEMMs on v_wmma_f32_16x16x32_bf16
// - Neighbor gather via global_load_async_to_lds_b128 (ASYNCcnt path)

#define B_SZ   4096
#define D_M    64
#define N_NB   64
#define HEADS  4

typedef __bf16 v16bf __attribute__((ext_vector_type(16)));
typedef __bf16 v8bf  __attribute__((ext_vector_type(8)));
typedef float  v8f   __attribute__((ext_vector_type(8)));

__device__ __forceinline__ v16bf load_frag(const __bf16* p0, const __bf16* p1) {
  union { v16bf v; v8bf h[2]; } x;
  x.h[0] = *(const v8bf*)p0;   // ds_load_b128
  x.h[1] = *(const v8bf*)p1;   // ds_load_b128
  return x.v;
}

// CDNA5 async global->LDS copy (16 B per lane). vdst = LDS byte offset VGPR,
// vaddr = 64-bit global address. Tracked by ASYNCcnt.
__device__ __forceinline__ void async_copy_b128(unsigned int lds_off, const void* gaddr) {
  asm volatile("global_load_async_to_lds_b128 %0, %1, off"
               :: "v"(lds_off), "v"(gaddr)
               : "memory");
}
__device__ __forceinline__ void wait_async0() {
  asm volatile("s_wait_asynccnt 0" ::: "memory");
}

// ---------------------------------------------------------------------------
// u[b,d] = user_emb[user_idx[b], d]
__global__ void gather_u_kernel(const int* __restrict__ uidx,
                                const float* __restrict__ user_emb,
                                float* __restrict__ u) {
  int i = blockIdx.x * blockDim.x + threadIdx.x;    // over B*64
  int b = i >> 6, d = i & 63;
  u[i] = user_emb[(size_t)uidx[b] * D_M + d];
}

// combined[b, 0:64] = u[b, :]
__global__ void copy_u_kernel(const float* __restrict__ u,
                              float* __restrict__ combined) {
  int i = blockIdx.x * blockDim.x + threadIdx.x;    // over B*64
  int b = i >> 6, d = i & 63;
  combined[(size_t)b * 192 + d] = u[i];
}

// ---------------------------------------------------------------------------
// One block (4 wave32) = one batch element of one MHA branch.
__global__ void __launch_bounds__(128)
mha_kernel(const float* __restrict__ u,        // [B,64] current user embeds
           const float* __restrict__ table,    // embedding table rows [*,64]
           const int*   __restrict__ neigh,    // [B,64]
           const float* __restrict__ W,        // [4,64,64] this layer (q,k,v,o)
           const float* __restrict__ bias4,    // [4,64]
           float* __restrict__ combined,       // [B,192]; write cols col0..+63
           int col0) {
  constexpr int LDA = 72;   // bf16 halves; 144 B row stride (16-B aligned)
  __shared__ __align__(16) float  stage[N_NB * D_M];   // async landing, f32
  __shared__ __align__(16) __bf16 kvh[N_NB * LDA];     // neighbor rows, bf16
  __shared__ __align__(16) __bf16 wT [D_M  * LDA];     // W^T (n-major), bf16
  __shared__ __align__(16) __bf16 Kb [N_NB * LDA];     // K results, bf16
  __shared__ __align__(16) __bf16 Vb [N_NB * LDA];     // V results, bf16
  __shared__ float qv[D_M];
  __shared__ float Qp[D_M];
  __shared__ float sc[HEADS * N_NB];
  __shared__ float attv[D_M];

  const int b    = blockIdx.x;
  const int tid  = threadIdx.x;
  const int wave = tid >> 5, lane = tid & 31;
  const int hl   = lane >> 4, r16 = lane & 15;

  if (tid == 0) {                         // warm weights (global_prefetch_b8)
    __builtin_prefetch(W, 0, 3);
    __builtin_prefetch(W + 2 * D_M * D_M, 0, 3);
  }

  // --- issue async DMA of the 64 neighbor rows into LDS staging -----------
  // 1024 x b128 chunks: 8 rounds x 128 lanes, 16 B per lane.
#pragma unroll
  for (int it = 0; it < 8; ++it) {
    int c  = tid + it * 128;          // chunk id: row n = c/16, 16B chunk c%16
    int n  = c >> 4;
    int cw = c & 15;
    int row = neigh[b * N_NB + n];
    const float* g = table + (size_t)row * D_M + cw * 4;
    async_copy_b128((unsigned int)(uintptr_t)&stage[c * 4], g);
  }

  if (tid < D_M) qv[tid] = u[(size_t)b * D_M + tid];

  // Wk transposed -> LDS bf16 (normal VMEM path, overlaps the async DMA)
  for (int idx = tid; idx < D_M * D_M; idx += blockDim.x) {
    int k = idx >> 6, n = idx & 63;
    wT[n * LDA + k] = (__bf16)W[1 * D_M * D_M + k * D_M + n];
  }

  wait_async0();            // s_wait_asynccnt 0: staging tile landed
  __syncthreads();

  // staging f32 -> bf16 A-operand tile
  for (int idx = tid; idx < N_NB * D_M; idx += blockDim.x) {
    int n = idx >> 6, d = idx & 63;
    kvh[n * LDA + d] = (__bf16)stage[idx];
  }
  // Q projection (1 row -> scalar path; rows of W are coalesced)
  if (tid < D_M) {
    float acc = bias4[0 * D_M + tid];
    for (int i = 0; i < D_M; ++i) acc += qv[i] * W[0 * D_M * D_M + i * D_M + tid];
    Qp[tid] = acc;
  }
  __syncthreads();

  // ---- K = kv @ Wk  (WMMA, wave w owns rows m0..m0+15) ----
  {
    const int m0 = wave * 16;
    for (int n0 = 0; n0 < D_M; n0 += 16) {
      v8f acc = {};
#pragma unroll
      for (int k0 = 0; k0 < D_M; k0 += 32) {
        const __bf16* ap = &kvh[(m0 + r16) * LDA + k0 + hl * 8];
        const __bf16* bp = &wT [(n0 + r16) * LDA + k0 + hl * 8];
        v16bf a = load_frag(ap, ap + 16);
        v16bf bb = load_frag(bp, bp + 16);
        acc = __builtin_amdgcn_wmma_f32_16x16x32_bf16(false, a, false, bb,
                                                      (short)0, acc, false, false);
      }
      float bv = bias4[1 * D_M + n0 + r16];
      int mb = hl * 8;
#pragma unroll
      for (int rr = 0; rr < 8; ++rr)
        Kb[(m0 + mb + rr) * LDA + n0 + r16] = (__bf16)(acc[rr] + bv);
    }
  }
  __syncthreads();
  // Wv transposed -> LDS (reuse wT)
  for (int idx = tid; idx < D_M * D_M; idx += blockDim.x) {
    int k = idx >> 6, n = idx & 63;
    wT[n * LDA + k] = (__bf16)W[2 * D_M * D_M + k * D_M + n];
  }
  __syncthreads();
  // ---- V = kv @ Wv (WMMA) ----
  {
    const int m0 = wave * 16;
    for (int n0 = 0; n0 < D_M; n0 += 16) {
      v8f acc = {};
#pragma unroll
      for (int k0 = 0; k0 < D_M; k0 += 32) {
        const __bf16* ap = &kvh[(m0 + r16) * LDA + k0 + hl * 8];
        const __bf16* bp = &wT [(n0 + r16) * LDA + k0 + hl * 8];
        v16bf a = load_frag(ap, ap + 16);
        v16bf bb = load_frag(bp, bp + 16);
        acc = __builtin_amdgcn_wmma_f32_16x16x32_bf16(false, a, false, bb,
                                                      (short)0, acc, false, false);
      }
      float bv = bias4[2 * D_M + n0 + r16];
      int mb = hl * 8;
#pragma unroll
      for (int rr = 0; rr < 8; ++rr)
        Vb[(m0 + mb + rr) * LDA + n0 + r16] = (__bf16)(acc[rr] + bv);
    }
  }
  __syncthreads();

  // scores[h,n] = (Q_h . K_{n,h}) / sqrt(16)
  for (int idx = tid; idx < HEADS * N_NB; idx += blockDim.x) {
    int h = idx >> 6, n = idx & 63;
    float s = 0.f;
#pragma unroll
    for (int i = 0; i < 16; ++i)
      s += Qp[h * 16 + i] * (float)Kb[n * LDA + h * 16 + i];
    sc[h * N_NB + n] = s * 0.25f;
  }
  __syncthreads();
  // softmax over n: one wave per head, 2 elems/lane, wave32 shuffles
  {
    int h = wave;
    float v0 = sc[h * N_NB + lane], v1 = sc[h * N_NB + 32 + lane];
    float m = fmaxf(v0, v1);
    for (int off = 16; off > 0; off >>= 1) m = fmaxf(m, __shfl_xor(m, off, 32));
    float e0 = __expf(v0 - m), e1 = __expf(v1 - m);
    float s = e0 + e1;
    for (int off = 16; off > 0; off >>= 1) s += __shfl_xor(s, off, 32);
    float inv = 1.0f / s;
    sc[h * N_NB + lane] = e0 * inv;
    sc[h * N_NB + 32 + lane] = e1 * inv;
  }
  __syncthreads();
  // att[d] = sum_n attn[h(d),n] * V[n,d]
  if (tid < D_M) {
    int h = tid >> 4;
    float acc = 0.f;
    for (int n = 0; n < N_NB; ++n)
      acc += sc[h * N_NB + n] * (float)Vb[n * LDA + tid];
    attv[tid] = acc;
  }
  __syncthreads();
  // out = att @ Wo + bo  -> combined[:, col0:col0+64]
  if (tid < D_M) {
    float acc = bias4[3 * D_M + tid];
    for (int i = 0; i < D_M; ++i) acc += attv[i] * W[3 * D_M * D_M + i * D_M + tid];
    combined[(size_t)b * 192 + col0 + tid] = acc;
  }
}

// ---------------------------------------------------------------------------
// Y[64 rows x 64] = act(X[64 rows x 192] @ Wf[192,64] + bias) via WMMA bf16.
__global__ void __launch_bounds__(128)
gemm192_kernel(const float* __restrict__ X,     // [B,192]
               const float* __restrict__ Wf,    // [192,64]
               const float* __restrict__ bias,  // [64]
               float* __restrict__ Y,           // [B,64]
               int do_relu) {
  constexpr int K = 192, Nc = 64, LDX = 208;   // 416 B stride, 16-B aligned
  __shared__ __align__(16) __bf16 Xs[64 * LDX];
  __shared__ __align__(16) __bf16 Ws[Nc * LDX];

  const int row0 = blockIdx.x * 64;
  const int tid  = threadIdx.x;
  if (tid == 0) __builtin_prefetch(Wf, 0, 3);

  for (int idx = tid; idx < 64 * K; idx += blockDim.x) {
    int rr = idx / K, c = idx - rr * K;
    Xs[rr * LDX + c] = (__bf16)X[(size_t)(row0 + rr) * K + c];
  }
  for (int idx = tid; idx < K * Nc; idx += blockDim.x) {
    int k = idx >> 6, n = idx & 63;
    Ws[n * LDX + k] = (__bf16)Wf[k * Nc + n];
  }
  __syncthreads();

  const int wave = tid >> 5, lane = tid & 31;
  const int hl = lane >> 4, r16 = lane & 15;
  const int m0 = wave * 16;
  for (int n0 = 0; n0 < Nc; n0 += 16) {
    v8f acc = {};
#pragma unroll
    for (int k0 = 0; k0 < K; k0 += 32) {
      const __bf16* ap = &Xs[(m0 + r16) * LDX + k0 + hl * 8];
      const __bf16* bp = &Ws[(n0 + r16) * LDX + k0 + hl * 8];
      v16bf a = load_frag(ap, ap + 16);
      v16bf bb = load_frag(bp, bp + 16);
      acc = __builtin_amdgcn_wmma_f32_16x16x32_bf16(false, a, false, bb,
                                                    (short)0, acc, false, false);
    }
    float bv = bias[n0 + r16];
    int mb = hl * 8;
#pragma unroll
    for (int rr = 0; rr < 8; ++rr) {
      float v = acc[rr] + bv;
      if (do_relu) v = fmaxf(v, 0.f);
      Y[(size_t)(row0 + m0 + mb + rr) * Nc + n0 + r16] = v;
    }
  }
}

// ---------------------------------------------------------------------------
// ratings + helpfulness head tail. One block (64 thr) per batch element.
__global__ void __launch_bounds__(64)
final_kernel(const float* __restrict__ u,      // [B,64] final user embeds
             const float* __restrict__ h1,     // [B,64] relu(combined@W1+b1)
             const int* __restrict__ uidx, const int* __restrict__ pidx,
             const int* __restrict__ cidx,
             const float* __restrict__ prod_emb, const float* __restrict__ cat_emb,
             const float* __restrict__ ubias, const float* __restrict__ pbias,
             const float* __restrict__ gbias,
             const float* __restrict__ W2, const float* __restrict__ b2,
             float* __restrict__ out, int batch) {
  __shared__ float p1[64];
  __shared__ float p2[64];
  int b = blockIdx.x, d = threadIdx.x;
  int pi = pidx[b], ci = cidx[b];
  float cp = prod_emb[(size_t)pi * D_M + d] + 0.3f * cat_emb[(size_t)ci * D_M + d];
  p1[d] = u[(size_t)b * D_M + d] * cp;
  p2[d] = h1[(size_t)b * D_M + d] * W2[d];
  __syncthreads();
  for (int off = 32; off > 0; off >>= 1) {
    if (d < off) { p1[d] += p1[d + off]; p2[d] += p2[d + off]; }
    __syncthreads();
  }
  if (d == 0) {
    out[b] = gbias[0] + ubias[uidx[b]] + pbias[pi] + p1[0];
    float h = p2[0] + b2[0];
    out[batch + b] = 1.0f / (1.0f + __expf(-h));
  }
}

// ---------------------------------------------------------------------------
extern "C" void kernel_launch(void* const* d_in, const int* in_sizes, int n_in,
                              void* d_out, int out_size, void* d_ws, size_t ws_size,
                              hipStream_t stream) {
  (void)in_sizes; (void)n_in; (void)out_size; (void)ws_size;
  const int*   user_idx  = (const int*)  d_in[0];
  const int*   prod_idx  = (const int*)  d_in[1];
  const int*   cat_idx   = (const int*)  d_in[2];
  const int*   soc_nb    = (const int*)  d_in[3];
  const int*   prd_nb    = (const int*)  d_in[4];
  const float* user_emb  = (const float*)d_in[5];
  const float* prod_emb  = (const float*)d_in[6];
  const float* cat_emb   = (const float*)d_in[7];
  const float* ubias_tab = (const float*)d_in[8];
  const float* pbias_tab = (const float*)d_in[9];
  const float* gbias     = (const float*)d_in[10];
  const float* soc_W     = (const float*)d_in[11];
  const float* soc_b     = (const float*)d_in[12];
  const float* int_W     = (const float*)d_in[13];
  const float* int_b     = (const float*)d_in[14];
  const float* fus_W     = (const float*)d_in[15];
  const float* fus_b     = (const float*)d_in[16];
  const float* help_W1   = (const float*)d_in[17];
  const float* help_b1   = (const float*)d_in[18];
  const float* help_W2   = (const float*)d_in[19];
  const float* help_b2   = (const float*)d_in[20];

  float* u        = (float*)d_ws;                  // [B,64]
  float* combined = u + (size_t)B_SZ * 64;         // [B,192]
  float* h1       = combined + (size_t)B_SZ * 192; // [B,64]
  float* out      = (float*)d_out;                 // [2*B]: ratings, helpfulness

  const int elems = B_SZ * 64;
  gather_u_kernel<<<elems / 256, 256, 0, stream>>>(user_idx, user_emb, u);

  for (int l = 0; l < 2; ++l) {
    copy_u_kernel<<<elems / 256, 256, 0, stream>>>(u, combined);
    mha_kernel<<<B_SZ, 128, 0, stream>>>(u, user_emb, soc_nb,
                                         soc_W + (size_t)l * 4 * 64 * 64,
                                         soc_b + (size_t)l * 4 * 64,
                                         combined, 64);
    mha_kernel<<<B_SZ, 128, 0, stream>>>(u, prod_emb, prd_nb,
                                         int_W + (size_t)l * 4 * 64 * 64,
                                         int_b + (size_t)l * 4 * 64,
                                         combined, 128);
    gemm192_kernel<<<B_SZ / 64, 128, 0, stream>>>(combined,
                                                  fus_W + (size_t)l * 192 * 64,
                                                  fus_b + (size_t)l * 64,
                                                  u, 1);
  }
  // helpfulness hidden layer: relu(combined @ help_W1 + help_b1)
  gemm192_kernel<<<B_SZ / 64, 128, 0, stream>>>(combined, help_W1, help_b1, h1, 1);

  final_kernel<<<B_SZ, 64, 0, stream>>>(u, h1, user_idx, prod_idx, cat_idx,
                                        prod_emb, cat_emb, ubias_tab, pbias_tab,
                                        gbias, help_W2, help_b2, out, B_SZ);
}
